// CustomMultiheadAttention_85143431676311
// MI455X (gfx1250) — matrix operand order repaired and checked by reading, not compiled, not run on hardware
//
#include <hip/hip_runtime.h>

#define SEQ   2048
#define BATCH 2
#define EMB   1024
#define NHEAD 16
#define DHEAD 64
#define MROWS (SEQ*BATCH)   // 4096

typedef __attribute__((ext_vector_type(16))) __bf16 v16bf;
typedef __attribute__((ext_vector_type(8)))  __bf16 v8bf;
typedef __attribute__((ext_vector_type(4)))  __bf16 v4bf;
typedef __attribute__((ext_vector_type(8)))  float  v8f;
typedef __attribute__((ext_vector_type(4)))  float  v4f;

// A-fragment (16x32 MxK bf16): lane holds 16 contiguous K at offset 16*half
__device__ __forceinline__ v16bf ldA(const __bf16* row, int half) {
  v8bf lo = *(const v8bf*)(row + 16 * half);
  v8bf hi = *(const v8bf*)(row + 16 * half + 8);
  v16bf r;
#pragma unroll
  for (int i = 0; i < 8; ++i) { r[i] = lo[i]; r[i + 8] = hi[i]; }
  return r;
}
// B-fragment (32x16 KxN bf16): lane holds 16 contiguous K at offset 16*half
__device__ __forceinline__ v16bf ldB(const __bf16* p) {
  v8bf lo = *(const v8bf*)(p);
  v8bf hi = *(const v8bf*)(p + 8);
  v16bf r;
#pragma unroll
  for (int i = 0; i < 8; ++i) { r[i] = lo[i]; r[i + 8] = hi[i]; }
  return r;
}
__device__ __forceinline__ v8f wmma_bf16(v16bf a, v16bf b, v8f c) {
  return __builtin_amdgcn_wmma_f32_16x16x32_bf16(false, a, false, b, (short)0, c,
                                                 false, false);
}
__device__ __forceinline__ v8f v8f_zero() {
  v8f z = {0.f, 0.f, 0.f, 0.f, 0.f, 0.f, 0.f, 0.f};
  return z;
}
__device__ __forceinline__ v4bf cvt4(v4f v) {
  v4bf p = {(__bf16)v[0], (__bf16)v[1], (__bf16)v[2], (__bf16)v[3]};
  return p;
}

// ---------------------------------------------------------------------------
// Kernel 1: Q/K/V projections  Y = X @ W^T + b, stored bf16.
// grid (64, 8, 3), block 256. Tile 64(M) x 128(N), 8 waves (2x4), K-step 64.
// ---------------------------------------------------------------------------
__global__ __launch_bounds__(256) void qkv_proj_kernel(
    const float* __restrict__ q_in, const float* __restrict__ k_in,
    const float* __restrict__ v_in, const float* __restrict__ Wq,
    const float* __restrict__ bq, const float* __restrict__ Wk,
    const float* __restrict__ bk, const float* __restrict__ Wv,
    const float* __restrict__ bv, __bf16* __restrict__ Yq,
    __bf16* __restrict__ Yk, __bf16* __restrict__ Yv) {
  const int z = blockIdx.z;
  const float* __restrict__ X  = (z == 0) ? q_in : (z == 1) ? k_in : v_in;
  const float* __restrict__ W  = (z == 0) ? Wq   : (z == 1) ? Wk   : Wv;
  const float* __restrict__ bs = (z == 0) ? bq   : (z == 1) ? bk   : bv;
  __bf16* __restrict__ Y       = (z == 0) ? Yq   : (z == 1) ? Yk   : Yv;

  const int m0 = blockIdx.x * 64;
  const int n0 = blockIdx.y * 128;

  __shared__ __align__(32) __bf16 As[64 * 64];   //  8 KB
  __shared__ __align__(32) __bf16 Bs[128 * 64];  // 16 KB

  const int tid = threadIdx.x;
  const int wave = tid >> 5, lane = tid & 31;
  const int wm = wave >> 2, wn = wave & 3;
  const int half = lane >> 4, l16 = lane & 15;

  v8f acc[2][2];
#pragma unroll
  for (int i = 0; i < 2; ++i)
#pragma unroll
    for (int j = 0; j < 2; ++j) acc[i][j] = v8f_zero();

  for (int k0 = 0; k0 < EMB; k0 += 64) {
    // Stage A (64x64) and B (128x64): float4 loads -> packed bf16 ds_store_b64
#pragma unroll
    for (int i = 0; i < 4; ++i) {
      int q = i * 256 + tid;            // quad index, 1024 quads
      int r = q >> 4, c = (q & 15) * 4;
      v4f v = *(const v4f*)&X[(size_t)(m0 + r) * EMB + k0 + c];
      *(v4bf*)&As[r * 64 + c] = cvt4(v);
    }
#pragma unroll
    for (int i = 0; i < 8; ++i) {
      int q = i * 256 + tid;            // 2048 quads
      int r = q >> 4, c = (q & 15) * 4;
      v4f v = *(const v4f*)&W[(size_t)(n0 + r) * EMB + k0 + c];
      *(v4bf*)&Bs[r * 64 + c] = cvt4(v);
    }
    __syncthreads();
    if (k0 + 64 < EMB) {  // global_prefetch_b8 of next K tile
      __builtin_prefetch(&X[(size_t)(m0 + (tid >> 2)) * EMB + k0 + 64], 0, 1);
      __builtin_prefetch(&W[(size_t)(n0 + (tid >> 1)) * EMB + k0 + 64], 0, 1);
    }
    v16bf af[2][2], bf_[2][2];
#pragma unroll
    for (int mt = 0; mt < 2; ++mt)
#pragma unroll
      for (int ks = 0; ks < 2; ++ks)
        af[mt][ks] = ldA(&As[(wm * 32 + mt * 16 + l16) * 64 + ks * 32], half);
#pragma unroll
    for (int nt = 0; nt < 2; ++nt)
#pragma unroll
      for (int ks = 0; ks < 2; ++ks)
        bf_[nt][ks] =
            ldB(&Bs[(wn * 32 + nt * 16 + l16) * 64 + ks * 32 + 16 * half]);
#pragma unroll
    for (int mt = 0; mt < 2; ++mt)
#pragma unroll
      for (int nt = 0; nt < 2; ++nt) {
        acc[mt][nt] = wmma_bf16(af[mt][0], bf_[nt][0], acc[mt][nt]);
        acc[mt][nt] = wmma_bf16(af[mt][1], bf_[nt][1], acc[mt][nt]);
      }
    __syncthreads();
  }

#pragma unroll
  for (int mt = 0; mt < 2; ++mt)
#pragma unroll
    for (int nt = 0; nt < 2; ++nt) {
      int n = n0 + wn * 32 + nt * 16 + l16;
      float bb = bs[n];
#pragma unroll
      for (int j = 0; j < 8; ++j) {
        int m = m0 + wm * 32 + mt * 16 + j + 8 * half;
        Y[(size_t)m * EMB + n] = (__bf16)(acc[mt][nt][j] + bb);
      }
    }
}

// ---------------------------------------------------------------------------
// Kernel 2: scores = QK^T/8 (+bias in softmax), softmax, write weights + diag.
// grid (64, 32) = (S/32, H*B), block 256, dyn LDS: 32x2048 f32 + 32x64 bf16.
// Q-block staged via GLOBAL_LOAD_ASYNC_TO_LDS_B128 (ASYNCcnt path).
// ---------------------------------------------------------------------------
__global__ __launch_bounds__(256) void attn_softmax_kernel(
    const __bf16* __restrict__ Yq, const __bf16* __restrict__ Yk,
    const float* __restrict__ abias, float* __restrict__ attn,
    float* __restrict__ diag) {
  extern __shared__ char smem_raw[];
  float* sc = (float*)smem_raw;                                         // 32xSEQ
  __bf16* qs = (__bf16*)(smem_raw + (size_t)32 * SEQ * sizeof(float));  // 32x64

  const int s0 = blockIdx.x * 32;
  const int hb = blockIdx.y;
  const int h = hb >> 1, b = hb & 1;

  const int tid = threadIdx.x, wave = tid >> 5, lane = tid & 31;
  const int half = lane >> 4, l16 = lane & 15;

  // Async-stage Q block (32 rows x 64 dims bf16 = 4KB): 16B per lane.
  {
    int r = tid >> 3, c = (tid & 7) * 8;
    const __bf16* gsrc =
        &Yq[((size_t)(s0 + r) * BATCH + b) * EMB + h * DHEAD + c];
    unsigned lds_off = (unsigned)(size_t)(void*)&qs[r * 64 + c];
    asm volatile("global_load_async_to_lds_b128 %0, %1, off" ::"v"(lds_off),
                 "v"(gsrc)
                 : "memory");
    asm volatile("s_wait_asynccnt 0x0" ::: "memory");
  }
  __syncthreads();

  // A fragments live in registers for the whole t-loop
  v16bf af[2][2];
#pragma unroll
  for (int mt = 0; mt < 2; ++mt)
#pragma unroll
    for (int ks = 0; ks < 2; ++ks)
      af[mt][ks] = ldA(&qs[(mt * 16 + l16) * 64 + ks * 32], half);

  for (int ti = wave; ti < SEQ / 16; ti += 8) {
    const int t0 = ti * 16;
    v16bf bf_[2];
#pragma unroll
    for (int ks = 0; ks < 2; ++ks)
      bf_[ks] = ldB(&Yk[((size_t)(t0 + l16) * BATCH + b) * EMB + h * DHEAD +
                        ks * 32 + 16 * half]);
#pragma unroll
    for (int mt = 0; mt < 2; ++mt) {
      v8f acc = v8f_zero();
      acc = wmma_bf16(af[mt][0], bf_[0], acc);
      acc = wmma_bf16(af[mt][1], bf_[1], acc);
#pragma unroll
      for (int j = 0; j < 8; ++j) {
        int r = mt * 16 + j + 8 * half;
        sc[r * SEQ + t0 + l16] = acc[j] * 0.125f;
      }
    }
  }
  __syncthreads();

  // Softmax: each wave owns 4 rows; float4-vectorized, bias fused in pass 1.
#pragma unroll
  for (int rr = 0; rr < 4; ++rr) {
    int r = wave * 4 + rr;
    int sg = s0 + r;
    v4f* row4 = (v4f*)&sc[r * SEQ];
    const v4f* bias4 = (const v4f*)&abias[(size_t)sg * SEQ];
    float mx = -3.402823466e38f;
    for (int i = lane; i < SEQ / 4; i += 32) {
      v4f v = row4[i] + bias4[i];
      row4[i] = v;
      mx = fmaxf(mx, fmaxf(fmaxf(v[0], v[1]), fmaxf(v[2], v[3])));
    }
#pragma unroll
    for (int off = 16; off > 0; off >>= 1)
      mx = fmaxf(mx, __shfl_xor(mx, off, 32));
    float sum = 0.f;
    for (int i = lane; i < SEQ / 4; i += 32) {
      v4f v = row4[i];
      v4f e;
      e[0] = __expf(v[0] - mx);
      e[1] = __expf(v[1] - mx);
      e[2] = __expf(v[2] - mx);
      e[3] = __expf(v[3] - mx);
      row4[i] = e;
      sum += e[0] + e[1] + e[2] + e[3];
    }
#pragma unroll
    for (int off = 16; off > 0; off >>= 1) sum += __shfl_xor(sum, off, 32);
    float inv = 1.0f / sum;
    v4f* dst4 = (v4f*)&attn[((size_t)hb * SEQ + sg) * SEQ];
    for (int i = lane; i < SEQ / 4; i += 32) {
      v4f v = row4[i] * inv;
      dst4[i] = v;
      if ((sg >> 2) == i) diag[(size_t)hb * SEQ + sg] = v[sg & 3];
    }
  }
}

// ---------------------------------------------------------------------------
// Kernel 3: out = (diag .* V) @ Wo^T + bo   (fp32 output)
// grid (64, 8), block 256. Same tiling as kernel 1.
// ---------------------------------------------------------------------------
__global__ __launch_bounds__(256) void out_proj_kernel(
    const __bf16* __restrict__ Yv, const float* __restrict__ diag,
    const float* __restrict__ Wo, const float* __restrict__ bo,
    float* __restrict__ out) {
  const int m0 = blockIdx.x * 64;
  const int n0 = blockIdx.y * 128;

  __shared__ __align__(32) __bf16 As[64 * 64];
  __shared__ __align__(32) __bf16 Bs[128 * 64];

  const int tid = threadIdx.x;
  const int wave = tid >> 5, lane = tid & 31;
  const int wm = wave >> 2, wn = wave & 3;
  const int half = lane >> 4, l16 = lane & 15;

  v8f acc[2][2];
#pragma unroll
  for (int i = 0; i < 2; ++i)
#pragma unroll
    for (int j = 0; j < 2; ++j) acc[i][j] = v8f_zero();

  for (int k0 = 0; k0 < EMB; k0 += 64) {
#pragma unroll
    for (int i = 0; i < 4; ++i) {
      int q = i * 256 + tid;
      int r = q >> 4, c = (q & 15) * 4;
      int m = m0 + r, e = k0 + c;  // quad never crosses a 64-wide head block
      float dv = diag[((size_t)((e >> 6) * BATCH + (m & 1))) * SEQ + (m >> 1)];
      v4bf src = *(const v4bf*)&Yv[(size_t)m * EMB + e];
      v4bf p = {(__bf16)((float)src[0] * dv), (__bf16)((float)src[1] * dv),
                (__bf16)((float)src[2] * dv), (__bf16)((float)src[3] * dv)};
      *(v4bf*)&As[r * 64 + c] = p;
    }
#pragma unroll
    for (int i = 0; i < 8; ++i) {
      int q = i * 256 + tid;
      int r = q >> 4, c = (q & 15) * 4;
      v4f v = *(const v4f*)&Wo[(size_t)(n0 + r) * EMB + k0 + c];
      *(v4bf*)&Bs[r * 64 + c] = cvt4(v);
    }
    __syncthreads();
    if (k0 + 64 < EMB)
      __builtin_prefetch(&Wo[(size_t)(n0 + (tid >> 1)) * EMB + k0 + 64], 0, 1);
    v16bf af[2][2], bf_[2][2];
#pragma unroll
    for (int mt = 0; mt < 2; ++mt)
#pragma unroll
      for (int ks = 0; ks < 2; ++ks)
        af[mt][ks] = ldA(&As[(wm * 32 + mt * 16 + l16) * 64 + ks * 32], half);
#pragma unroll
    for (int nt = 0; nt < 2; ++nt)
#pragma unroll
      for (int ks = 0; ks < 2; ++ks)
        bf_[nt][ks] =
            ldB(&Bs[(wn * 32 + nt * 16 + l16) * 64 + ks * 32 + 16 * half]);
#pragma unroll
    for (int mt = 0; mt < 2; ++mt)
#pragma unroll
      for (int nt = 0; nt < 2; ++nt) {
        acc[mt][nt] = wmma_bf16(af[mt][0], bf_[nt][0], acc[mt][nt]);
        acc[mt][nt] = wmma_bf16(af[mt][1], bf_[nt][1], acc[mt][nt]);
      }
    __syncthreads();
  }

#pragma unroll
  for (int mt = 0; mt < 2; ++mt)
#pragma unroll
    for (int nt = 0; nt < 2; ++nt) {
      int n = n0 + wn * 32 + nt * 16 + l16;
      float bb = bo[n];
#pragma unroll
      for (int j = 0; j < 8; ++j) {
        int m = m0 + wm * 32 + mt * 16 + j + 8 * half;
        out[(size_t)m * EMB + n] = acc[mt][nt][j] + bb;
      }
    }
}

// ---------------------------------------------------------------------------
extern "C" void kernel_launch(void* const* d_in, const int* in_sizes, int n_in,
                              void* d_out, int out_size, void* d_ws,
                              size_t ws_size, hipStream_t stream) {
  const float* q_in  = (const float*)d_in[0];
  const float* k_in  = (const float*)d_in[1];
  const float* v_in  = (const float*)d_in[2];
  const float* abias = (const float*)d_in[3];
  const float* Wq = (const float*)d_in[4];
  const float* bq = (const float*)d_in[5];
  const float* Wk = (const float*)d_in[6];
  const float* bk = (const float*)d_in[7];
  const float* Wv = (const float*)d_in[8];
  const float* bv = (const float*)d_in[9];
  const float* Wo = (const float*)d_in[10];
  const float* bo = (const float*)d_in[11];

  // d_out: [ out (S,B,E) fp32 | attn_weights (H,B,S,S) fp32 ]
  float* out  = (float*)d_out;
  float* attn = out + (size_t)MROWS * EMB;

  // Workspace: bf16 Q, K, V projections + fp32 diagonal
  __bf16* Yq = (__bf16*)d_ws;
  __bf16* Yk = Yq + (size_t)MROWS * EMB;
  __bf16* Yv = Yk + (size_t)MROWS * EMB;
  float* diag = (float*)(Yv + (size_t)MROWS * EMB);  // NHEAD*BATCH*SEQ floats

  qkv_proj_kernel<<<dim3(MROWS / 64, EMB / 128, 3), 256, 0, stream>>>(
      q_in, k_in, v_in, Wq, bq, Wk, bk, Wv, bv, Yq, Yk, Yv);

  const size_t attn_smem =
      (size_t)32 * SEQ * sizeof(float) + (size_t)32 * DHEAD * sizeof(__bf16);
  attn_softmax_kernel<<<dim3(SEQ / 32, NHEAD * BATCH), 256, attn_smem,
                        stream>>>(Yq, Yk, abias, attn, diag);

  out_proj_kernel<<<dim3(MROWS / 64, EMB / 128), 256, 0, stream>>>(
      Yv, diag, Wo, bo, out);
}
